// LConvBilinShift_43800076485326
// MI455X (gfx1250) — matrix operand order repaired and checked by reading, not compiled, not run on hardware
//
#include <hip/hip_runtime.h>

typedef __attribute__((ext_vector_type(16))) _Float16 v16h;
typedef __attribute__((ext_vector_type(8)))  float    v8f;

#define XDIM 256
#define YDIM 256
#define CCH  4
// A-fragment cache: [var(2)][mt(5)][ks(3)][lane(32)][16 f16]
#define WS_A_HALFS (2 * 5 * 3 * 32 * 16)
#define WS_A_BYTES (WS_A_HALFS * 2)

__device__ __forceinline__ void cmul(float ar, float ai, float br, float bi,
                                     float& cr, float& ci) {
  cr = ar * br - ai * bi;
  ci = ar * bi + ai * br;
}

// A_re = [Re(alpha) | -Im(alpha)], A_im = [Im(alpha) | Re(alpha)], zero-padded
__device__ __forceinline__ float alpha_frag_val(const float* __restrict__ alpha,
                                                int var, int m, int kg) {
  if (m >= 72 || kg >= 82) return 0.f;
  int kk = (kg < 41) ? kg : kg - 41;
  const float* a = alpha + 2 * ((size_t)m * 41 + kk);
  float ar = a[0], ai = a[1];
  if (var == 0) return (kg < 41) ? ar : -ai;
  return (kg < 41) ? ai : ar;
}

// One-shot: build A fragments in per-lane WMMA layout into workspace.
__global__ void prep_alpha_frags(const float* __restrict__ alpha,
                                 _Float16* __restrict__ wsA) {
  int e = blockIdx.x * blockDim.x + threadIdx.x;   // 960 fragment-lane entries
  if (e >= 2 * 5 * 3 * 32) return;
  int lane = e & 31;
  int ks   = (e >> 5) % 3;
  int mt   = (e / 96) % 5;
  int var  = e / 480;
  int m    = mt * 16 + (lane & 15);
  _Float16* dst = wsA + (size_t)e * 16;
  #pragma unroll
  for (int t = 0; t < 16; ++t) {
    // 16-bit A layout: lane<16 holds K {0-7,16-23}; lane>=16 holds {8-15,24-31}
    int klocal = ((t < 8) ? t : t + 8) + ((lane < 16) ? 0 : 8);
    int kg = ks * 32 + klocal;
    dst[t] = (_Float16)alpha_frag_val(alpha, var, m, kg);
  }
}

template <bool FROM_WS>
__launch_bounds__(256, 1)
__global__ void lconv_wmma_kernel(const float* __restrict__ plaqs,   // (B,4,X,Y) cfloat
                                  const float* __restrict__ links,   // (B,2,X,Y) cfloat
                                  const float* __restrict__ alpha,   // (8,9,41)  cfloat
                                  const _Float16* __restrict__ wsA,  // A frags (or null)
                                  float* __restrict__ out)           // (B,8,X,Y) cfloat
{
  __shared__ float sPlaq[3][CCH][YDIM][2];                 // 24576 B, f32 complex
  __shared__ float sLink[3][YDIM][2];                      //  6144 B
  __shared__ __align__(32) unsigned char sPool[8][3072];   // 24576 B: sT / sS union

  const int tid = threadIdx.x;
  const int w   = tid >> 5;
  const int l   = tid & 31;
  const int b   = blockIdx.x / XDIM;
  const int x   = blockIdx.x % XDIM;
  const int xr1 = (x + 1) & (XDIM - 1);
  const int xr2 = (x + 2) & (XDIM - 1);

  // ---- Phase 1: async DMA staging, global -> LDS (raw b128, ASYNCcnt) ----
  {
    const unsigned pl = (unsigned)(unsigned long long)(const void*)&sPlaq[0][0][0][0];
    const unsigned ll = (unsigned)(unsigned long long)(const void*)&sLink[0][0][0];
    const unsigned long long pbase = (unsigned long long)plaqs;
    const unsigned long long lbase = (unsigned long long)links;

    // plaq rows: 12 (r,c)-rows x 2048 B, 16 B per lane-transfer
    for (int idx = tid; idx < 1536; idx += 256) {
      int rr = idx >> 7;                 // 0..11 == r*4 + c
      int r  = rr >> 2;
      int c  = rr & 3;
      int rx = (r == 0) ? x : (r == 1 ? xr1 : xr2);
      unsigned within = (unsigned)(idx & 127) << 4;
      unsigned goff   = ((((unsigned)b * CCH + c) * XDIM + rx) * YDIM) * 8u + within;
      unsigned lds    = pl + (unsigned)rr * 2048u + within;
      asm volatile("global_load_async_to_lds_b128 %0, %1, %2"
                   :: "v"(lds), "v"(goff), "s"(pbase) : "memory");
    }
    // link rows: 3 rows x 2048 B (Ux@x, Ux@x+1, Uy@x)
    for (int idx = tid; idx < 384; idx += 256) {
      int rr = idx >> 7;                 // 0..2
      int d  = (rr == 2) ? 1 : 0;
      int rx = (rr == 1) ? xr1 : x;
      unsigned within = (unsigned)(idx & 127) << 4;
      unsigned goff   = ((((unsigned)b * 2 + d) * XDIM + rx) * YDIM) * 8u + within;
      unsigned lds    = ll + (unsigned)rr * 2048u + within;
      asm volatile("global_load_async_to_lds_b128 %0, %1, %2"
                   :: "v"(lds), "v"(goff), "s"(lbase) : "memory");
    }
    asm volatile("s_wait_asynccnt 0x0" ::: "memory");
  }
  __syncthreads();

  const int n    = l & 15;
  const int h    = l >> 4;
  const int hi8  = (l < 16) ? 0 : 8;
  const int ibas = (l < 16) ? 0 : 4;

  _Float16 (*sTw)[96]    = (_Float16(*)[96])(&sPool[w][0]);   // [16][96] f16, 3072 B
  float    (*sSw)[16][2] = (float(*)[16][2])(&sPool[w][0]);   // [16][16][2] f32, 2048 B

  // -------- Phase 2: build both 96x16 B operands; keep fragments in VGPRs --------
  v16h  bfrag[6];
  float wvr[2][4], wvi[2][4];

  for (int ng = 0; ng < 2; ++ng) {
    const int y  = ((w << 1) + ng) * 16 + n;
    const int y1 = (y + 1) & (YDIM - 1);
    const int y2 = (y + 2) & (YDIM - 1);

    for (int q = 0; q < 10; ++q) {
      int kk = h * 10 + q;
      int c  = kk & 3;
      int g  = kk >> 2;
      float tr, ti;
      if (g == 0) {
        tr = sPlaq[0][c][y][0]; ti = sPlaq[0][c][y][1];
      } else if (g == 1) {
        cmul(sLink[0][y][0], sLink[0][y][1],
             sPlaq[1][c][y][0], sPlaq[1][c][y][1], tr, ti);
      } else if (g == 2) {
        float ur, ui;
        cmul(sLink[0][y][0], sLink[0][y][1], sLink[1][y][0], sLink[1][y][1], ur, ui);
        cmul(ur, ui, sPlaq[2][c][y][0], sPlaq[2][c][y][1], tr, ti);
      } else if (g == 3) {
        cmul(sLink[2][y][0], sLink[2][y][1],
             sPlaq[0][c][y1][0], sPlaq[0][c][y1][1], tr, ti);
      } else {
        float ur, ui;
        cmul(sLink[2][y][0], sLink[2][y][1], sLink[2][y1][0], sLink[2][y1][1], ur, ui);
        cmul(ur, ui, sPlaq[0][c][y2][0], sPlaq[0][c][y2][1], tr, ti);
      }
      sTw[n][kk]      = (_Float16)tr;
      sTw[n][kk + 20] = (_Float16)tr;        // conj: Re unchanged
      sTw[n][kk + 41] = (_Float16)ti;
      sTw[n][kk + 61] = (_Float16)(-ti);     // conj: Im negated
    }
    if (h == 0) {
      sTw[n][40] = (_Float16)1.0f;
      sTw[n][81] = (_Float16)0.0f;
      for (int k = 82; k < 96; ++k) sTw[n][k] = (_Float16)0.0f;
    }
    asm volatile("s_wait_dscnt 0x0" ::: "memory");

    {
      const _Float16* base = &sTw[n][0];
      const int koff = (l < 16) ? 0 : 8;
      typedef __attribute__((ext_vector_type(8))) _Float16 v8h_;
      #pragma unroll
      for (int ks = 0; ks < 3; ++ks) {
        v8h_ lo = *(const v8h_*)(base + ks * 32 + koff);
        v8h_ hi = *(const v8h_*)(base + ks * 32 + 16 + koff);
        bfrag[ng * 3 + ks] = __builtin_shufflevector(lo, hi, 0,1,2,3,4,5,6,7,8,9,10,11,12,13,14,15);
      }
    }
    asm volatile("s_wait_dscnt 0x0" ::: "memory");  // B loads done before LDS reuse

    #pragma unroll
    for (int c = 0; c < 4; ++c) {
      wvr[ng][c] = sPlaq[0][c][y][0];
      wvi[ng][c] = sPlaq[0][c][y][1];
    }
  }

  float accr[2][4] = {{0.f,0.f,0.f,0.f},{0.f,0.f,0.f,0.f}};
  float acci[2][4] = {{0.f,0.f,0.f,0.f},{0.f,0.f,0.f,0.f}};

  // -------- Phase 3: mt outer; each A fragment feeds both site groups --------
  for (int mt = 0; mt < 5; ++mt) {
    v16h aRe[3], aIm[3];
    if (FROM_WS) {
      #pragma unroll
      for (int ks = 0; ks < 3; ++ks) {
        aRe[ks] = *(const v16h*)(wsA + (size_t)(((0 * 5 + mt) * 3 + ks) * 32 + l) * 16);
        aIm[ks] = *(const v16h*)(wsA + (size_t)(((1 * 5 + mt) * 3 + ks) * 32 + l) * 16);
      }
    } else {
      const int m = mt * 16 + (l & 15);
      #pragma unroll
      for (int ks = 0; ks < 3; ++ks) {
        #pragma unroll
        for (int t = 0; t < 16; ++t) {
          int klocal = ((t < 8) ? t : t + 8) + ((l < 16) ? 0 : 8);
          int kg = ks * 32 + klocal;
          aRe[ks][t] = (_Float16)alpha_frag_val(alpha, 0, m, kg);
          aIm[ks][t] = (_Float16)alpha_frag_val(alpha, 1, m, kg);
        }
      }
    }

    #pragma unroll
    for (int ng = 0; ng < 2; ++ng) {
      v8f cRe = {}; v8f cIm = {};
      cRe = __builtin_amdgcn_wmma_f32_16x16x32_f16(false, aRe[0], false, bfrag[ng*3+0], (short)0, cRe, false, false);
      cRe = __builtin_amdgcn_wmma_f32_16x16x32_f16(false, aRe[1], false, bfrag[ng*3+1], (short)0, cRe, false, false);
      cRe = __builtin_amdgcn_wmma_f32_16x16x32_f16(false, aRe[2], false, bfrag[ng*3+2], (short)0, cRe, false, false);
      cIm = __builtin_amdgcn_wmma_f32_16x16x32_f16(false, aIm[0], false, bfrag[ng*3+0], (short)0, cIm, false, false);
      cIm = __builtin_amdgcn_wmma_f32_16x16x32_f16(false, aIm[1], false, bfrag[ng*3+1], (short)0, cIm, false, false);
      cIm = __builtin_amdgcn_wmma_f32_16x16x32_f16(false, aIm[2], false, bfrag[ng*3+2], (short)0, cIm, false, false);

      #pragma unroll
      for (int v = 0; v < 8; ++v) {
        sSw[v + hi8][n][0] = cRe[v];
        sSw[v + hi8][n][1] = cIm[v];
      }
      asm volatile("s_wait_dscnt 0x0" ::: "memory");

      #pragma unroll
      for (int rl = 0; rl < 16; ++rl) {
        int mm = mt * 16 + rl;
        if (mm >= 72) break;
        int i = mm / 9;
        if (i < ibas || i >= ibas + 4) continue;
        int j = mm % 9;
        float sr = sSw[rl][n][0];
        float si = sSw[rl][n][1];
        float wr, wi;
        if (j == 8) { wr = 1.f; wi = 0.f; }
        else {
          int c = (j < 4) ? j : j - 4;
          wr = wvr[ng][c];
          wi = (j >= 4) ? -wvi[ng][c] : wvi[ng][c];
        }
        accr[ng][i - ibas] += sr * wr - si * wi;
        acci[ng][i - ibas] += sr * wi + si * wr;
      }
      asm volatile("s_wait_dscnt 0x0" ::: "memory");   // sS reads done before next write
    }
  }

  // -------- Phase 4: store 4 complex outputs per lane per site group --------
  #pragma unroll
  for (int ng = 0; ng < 2; ++ng) {
    const int y = ((w << 1) + ng) * 16 + n;
    #pragma unroll
    for (int q = 0; q < 4; ++q) {
      int i = ibas + q;
      size_t o = 2 * ((((size_t)b * 8 + i) * XDIM + x) * YDIM + y);
      out[o]     = accr[ng][q];
      out[o + 1] = acci[ng][q];
    }
  }
}

extern "C" void kernel_launch(void* const* d_in, const int* in_sizes, int n_in,
                              void* d_out, int out_size, void* d_ws, size_t ws_size,
                              hipStream_t stream) {
  const float* plaqs = (const float*)d_in[0];   // complex64 interleaved
  const float* links = (const float*)d_in[1];
  const float* alpha = (const float*)d_in[2];
  // d_in[3] = shift (==2, baked into channel construction)
  float* out = (float*)d_out;

  dim3 grid(8 * XDIM);
  dim3 block(256);

  if (d_ws != nullptr && ws_size >= (size_t)WS_A_BYTES) {
    _Float16* wsA = (_Float16*)d_ws;
    hipLaunchKernelGGL(prep_alpha_frags, dim3(4), dim3(256), 0, stream, alpha, wsA);
    hipLaunchKernelGGL(lconv_wmma_kernel<true>, grid, block, 0, stream,
                       plaqs, links, alpha, (const _Float16*)wsA, out);
  } else {
    hipLaunchKernelGGL(lconv_wmma_kernel<false>, grid, block, 0, stream,
                       plaqs, links, alpha, (const _Float16*)nullptr, out);
  }
}
